// GraphTransformerLayer_18236431139305
// MI455X (gfx1250) — matrix operand order repaired and checked by reading, not compiled, not run on hardware
//
#include <hip/hip_runtime.h>

// ---------------- problem constants (match reference) ----------------
enum { Bg = 4, Cdim = 128, Nn = 25000, Ee = 200000, Hh = 4, DHd = 32,
       OUTd = 128, EDd = 32, HIDd = 256 };
#define NB_NODES (Bg * Nn)   /* 100000 */
#define T_EDGES  (Bg * Ee)   /* 800000 */

// ---------------- WMMA types ----------------
typedef __attribute__((ext_vector_type(16))) __bf16 v16bf;
typedef __attribute__((ext_vector_type(8)))  float  v8f;
typedef __attribute__((ext_vector_type(4)))  unsigned int u32x4;

union FragBF { v16bf v; u32x4 q[2]; };

__device__ __forceinline__ unsigned short f2bf(float f) {
  unsigned u = __float_as_uint(f);
  u += 0x7FFFu + ((u >> 16) & 1u);     // round-to-nearest-even
  return (unsigned short)(u >> 16);
}

__device__ __forceinline__ unsigned fkey(float f) {   // order-preserving uint key
  unsigned u = __float_as_uint(f);
  return (u & 0x80000000u) ? ~u : (u | 0x80000000u);
}
__device__ __forceinline__ float fdec(unsigned k) {
  unsigned u = (k & 0x80000000u) ? (k ^ 0x80000000u) : ~k;
  return __uint_as_float(u);
}
__device__ __forceinline__ void atomAddF(float* p, float v) {
  __hip_atomic_fetch_add(p, v, __ATOMIC_RELAXED, __HIP_MEMORY_SCOPE_AGENT);
}

// ---------------- generic bf16-WMMA GEMM:  Y = act(X[MxK] * W[KxNc] + bias) ----------------
#define AST 40   // LDS row stride (shorts) for A tile (64 x 32)
#define BST 40   // LDS row stride (shorts) for B tile (128 x 32, stored transposed [n][k])

__global__ __launch_bounds__(256) void wmma_gemm_kernel(
    const float* __restrict__ X, const float* __restrict__ W,
    const float* __restrict__ bias, float* __restrict__ Y,
    int M, int K, int Nc, int relu)
{
  __shared__ alignas(16) unsigned short As[64 * AST];
  __shared__ alignas(16) unsigned short Bs[128 * BST];

  const int tid    = threadIdx.x;
  const int mblock = blockIdx.x * 64;
  const int nblock = blockIdx.y * 128;

  const int wave = tid >> 5;
  const int lane = tid & 31;
  const int hsel = lane >> 4;     // 0/1 : which 16-lane half
  const int l16  = lane & 15;
  const int mi   = wave & 3;      // 16-row subtile within 64
  const int nj   = wave >> 2;     // 64-col subtile (0/1)

  v8f acc[4];
  for (int j = 0; j < 4; ++j)
    for (int i = 0; i < 8; ++i) acc[j][i] = 0.0f;

  const int a_row = tid >> 2;          // 0..63
  const int a_k   = (tid & 3) * 8;     // 0,8,16,24
  const int b_k   = tid >> 3;          // 0..31
  const int b_n   = (tid & 7) * 16;    // 0..112

  for (int kb = 0; kb < K; kb += 32) {
    // stage A chunk (64 rows x 32 k), fp32 -> bf16
    {
      int grow = mblock + a_row;
      unsigned short* dp = &As[a_row * AST + a_k];
      if (grow < M) {
        const float* sp = X + (size_t)grow * K + kb + a_k;
        #pragma unroll
        for (int j = 0; j < 8; ++j) dp[j] = f2bf(sp[j]);
      } else {
        #pragma unroll
        for (int j = 0; j < 8; ++j) dp[j] = 0;
      }
    }
    // stage B chunk transposed: Bs[n][k]
    {
      const float* sp = W + (size_t)(kb + b_k) * Nc + nblock + b_n;
      #pragma unroll
      for (int j = 0; j < 16; ++j) Bs[(b_n + j) * BST + b_k] = f2bf(sp[j]);
    }
    __syncthreads();

    // A fragment: lane half=h -> elems 0..7 = K(h*8..h*8+7), elems 8..15 = K(16+h*8..)
    FragBF af;
    const int arow = mi * 16 + l16;
    af.q[0] = *(const u32x4*)&As[arow * AST + hsel * 8];
    af.q[1] = *(const u32x4*)&As[arow * AST + 16 + hsel * 8];

    // preload ALL four B fragments, then issue the WMMAs back-to-back
    // (single s_wait_dscnt, no per-WMMA LDS stall)
    FragBF bf[4];
    #pragma unroll
    for (int j = 0; j < 4; ++j) {
      const int ncol = nj * 64 + j * 16 + l16;
      bf[j].q[0] = *(const u32x4*)&Bs[ncol * BST + hsel * 8];
      bf[j].q[1] = *(const u32x4*)&Bs[ncol * BST + 16 + hsel * 8];
    }
    #pragma unroll
    for (int j = 0; j < 4; ++j) {
      acc[j] = __builtin_amdgcn_wmma_f32_16x16x32_bf16(
          false, af.v, false, bf[j].v, (short)0, acc[j], false, false);
    }
    __syncthreads();
  }

  #pragma unroll
  for (int j = 0; j < 4; ++j) {
    const int col = nblock + nj * 64 + j * 16 + l16;
    const float bv = bias ? bias[col] : 0.0f;
    #pragma unroll
    for (int r = 0; r < 8; ++r) {
      const int row = mblock + mi * 16 + hsel * 8 + r;
      if (row < M) {
        float vv = acc[j][r] + bv;
        if (relu) vv = fmaxf(vv, 0.0f);
        Y[(size_t)row * Nc + col] = vv;
      }
    }
  }
}

// ---------------- transposes ----------------
// x[B,C,N] -> xf[(b*N+n), C]
__global__ __launch_bounds__(256) void transpose_x_kernel(
    const float* __restrict__ x, float* __restrict__ xf)
{
  __shared__ float tile[32][33];
  const int bg = blockIdx.z, c0 = blockIdx.y * 32, n0 = blockIdx.x * 32;
  const int tx = threadIdx.x, ty = threadIdx.y;
  #pragma unroll
  for (int i = 0; i < 32; i += 8) {
    int n = n0 + tx;
    tile[ty + i][tx] = (n < Nn) ? x[((size_t)(bg * Cdim + c0 + ty + i)) * Nn + n] : 0.0f;
  }
  __syncthreads();
  #pragma unroll
  for (int i = 0; i < 32; i += 8) {
    int n = n0 + ty + i;
    if (n < Nn) xf[((size_t)(bg * Nn + n)) * Cdim + c0 + tx] = tile[tx][ty + i];
  }
}

// h[(b*N+n), OUT] -> out[B, OUT, N]
__global__ __launch_bounds__(256) void transpose_out_kernel(
    const float* __restrict__ h, float* __restrict__ out)
{
  __shared__ float tile[32][33];
  const int bg = blockIdx.z, o0 = blockIdx.y * 32, n0 = blockIdx.x * 32;
  const int tx = threadIdx.x, ty = threadIdx.y;
  #pragma unroll
  for (int i = 0; i < 32; i += 8) {
    int n = n0 + ty + i;
    tile[ty + i][tx] = (n < Nn) ? h[((size_t)(bg * Nn + n)) * OUTd + o0 + tx] : 0.0f;
  }
  __syncthreads();
  #pragma unroll
  for (int i = 0; i < 32; i += 8) {
    int n = n0 + tx;
    if (n < Nn) out[((size_t)(bg * OUTd + o0 + ty + i)) * Nn + n] = tile[tx][ty + i];
  }
}

// ---------------- edge pass A: alpha + running segment max ----------------
// one wave per edge; We staged in LDS; e_t = edge_attr[t] @ We recomputed on the fly
__global__ __launch_bounds__(256) void edge_alpha_kernel(
    const int* __restrict__ eidx, const float* __restrict__ eattr,
    const float* __restrict__ We, const float* __restrict__ q,
    const float* __restrict__ k, float* __restrict__ alpha,
    unsigned* __restrict__ keys)
{
  __shared__ float WeS[EDd * OUTd];     // 16 KB
  __shared__ float eaS[8][EDd];

  for (int i = threadIdx.x; i < EDd * OUTd; i += 256) WeS[i] = We[i];

  const int wave = threadIdx.x >> 5;
  const int lane = threadIdx.x & 31;
  const int t = blockIdx.x * 8 + wave;
  const bool valid = (t < T_EDGES);
  if (valid) eaS[wave][lane] = eattr[(size_t)t * EDd + lane];
  __syncthreads();
  if (!valid) return;

  const int b  = t / Ee;
  const int ei = t - b * Ee;
  const int src = eidx[(size_t)(b * 2) * Ee + ei];
  const int dst = eidx[(size_t)(b * 2 + 1) * Ee + ei];

  const float4 q4 = *(const float4*)&q[(size_t)dst * OUTd + lane * 4];
  const float4 k4 = *(const float4*)&k[(size_t)src * OUTd + lane * 4];

  float e0 = 0.f, e1 = 0.f, e2 = 0.f, e3 = 0.f;
  #pragma unroll 8
  for (int kk = 0; kk < EDd; ++kk) {
    const float a = eaS[wave][kk];
    const float4 w4 = *(const float4*)&WeS[kk * OUTd + lane * 4];
    e0 += a * w4.x; e1 += a * w4.y; e2 += a * w4.z; e3 += a * w4.w;
  }
  float s = q4.x * (k4.x + e0) + q4.y * (k4.y + e1)
          + q4.z * (k4.z + e2) + q4.w * (k4.w + e3);
  s += __shfl_xor(s, 1, 32);
  s += __shfl_xor(s, 2, 32);
  s += __shfl_xor(s, 4, 32);           // sum over the 8 lanes of one head
  if ((lane & 7) == 0) {
    const int h = lane >> 3;
    const float a = s * 0.17677669529663687f;   // 1/sqrt(32)
    alpha[(size_t)t * Hh + h] = a;
    atomicMax(&keys[(size_t)dst * Hh + h], fkey(a));
  }
}

// keys (uint) -> amax (float, isfinite rule), in place
__global__ void finalize_amax_kernel(unsigned* __restrict__ keys, int n)
{
  const int i = blockIdx.x * blockDim.x + threadIdx.x;
  if (i >= n) return;
  const unsigned kk = keys[i];
  float m = 0.0f;
  if (kk != 0u) { const float d = fdec(kk); if (__builtin_isfinite(d)) m = d; }
  ((float*)keys)[i] = m;
}

// ---------------- edge pass C: ea = exp(alpha - amax[dst]); denom += ea ----------------
__global__ void edge_softmax_kernel(
    const int* __restrict__ eidx, float* __restrict__ alpha,
    const float* __restrict__ amax, float* __restrict__ denom)
{
  const int i = blockIdx.x * blockDim.x + threadIdx.x;
  if (i >= T_EDGES * Hh) return;
  const int t = i >> 2, h = i & 3;
  const int b = t / Ee, ei = t - b * Ee;
  const int dst = eidx[(size_t)(b * 2 + 1) * Ee + ei];
  const float ea = expf(alpha[i] - amax[(size_t)dst * Hh + h]);
  alpha[i] = ea;
  atomAddF(&denom[(size_t)dst * Hh + h], ea);
}

// ---------------- edge pass D: out[dst] += attn * (v[src] + e_t) ----------------
__global__ __launch_bounds__(256) void edge_aggregate_kernel(
    const int* __restrict__ eidx, const float* __restrict__ eattr,
    const float* __restrict__ We, const float* __restrict__ v,
    const float* __restrict__ ea, const float* __restrict__ denom,
    float* __restrict__ outb)
{
  __shared__ float WeS[EDd * OUTd];
  __shared__ float eaS[8][EDd];

  for (int i = threadIdx.x; i < EDd * OUTd; i += 256) WeS[i] = We[i];

  const int wave = threadIdx.x >> 5;
  const int lane = threadIdx.x & 31;
  const int t = blockIdx.x * 8 + wave;
  const bool valid = (t < T_EDGES);
  if (valid) eaS[wave][lane] = eattr[(size_t)t * EDd + lane];
  __syncthreads();
  if (!valid) return;

  const int b  = t / Ee;
  const int ei = t - b * Ee;
  const int src = eidx[(size_t)(b * 2) * Ee + ei];
  const int dst = eidx[(size_t)(b * 2 + 1) * Ee + ei];

  const int h = lane >> 3;
  const float w = ea[(size_t)t * Hh + h] /
                  (denom[(size_t)dst * Hh + h] + 1e-16f);

  const float4 v4 = *(const float4*)&v[(size_t)src * OUTd + lane * 4];
  float e0 = 0.f, e1 = 0.f, e2 = 0.f, e3 = 0.f;
  #pragma unroll 8
  for (int kk = 0; kk < EDd; ++kk) {
    const float a = eaS[wave][kk];
    const float4 w4 = *(const float4*)&WeS[kk * OUTd + lane * 4];
    e0 += a * w4.x; e1 += a * w4.y; e2 += a * w4.z; e3 += a * w4.w;
  }
  float* op = &outb[(size_t)dst * OUTd + lane * 4];
  atomAddF(&op[0], (v4.x + e0) * w);
  atomAddF(&op[1], (v4.y + e1) * w);
  atomAddF(&op[2], (v4.z + e2) * w);
  atomAddF(&op[3], (v4.w + e3) * w);
}

// ---------------- host orchestration ----------------
extern "C" void kernel_launch(void* const* d_in, const int* in_sizes, int n_in,
                              void* d_out, int out_size, void* d_ws, size_t ws_size,
                              hipStream_t stream)
{
  (void)in_sizes; (void)n_in; (void)out_size; (void)ws_size;

  const float* x     = (const float*)d_in[0];
  const int*   eidx  = (const int*)  d_in[1];
  const float* eattr = (const float*)d_in[2];
  const float* Wq    = (const float*)d_in[3];  const float* bq = (const float*)d_in[4];
  const float* Wk    = (const float*)d_in[5];  const float* bk = (const float*)d_in[6];
  const float* Wv    = (const float*)d_in[7];  const float* bv = (const float*)d_in[8];
  const float* We    = (const float*)d_in[9];
  const float* Wskip = (const float*)d_in[10]; const float* bskip = (const float*)d_in[11];
  const float* W1    = (const float*)d_in[12]; const float* b1 = (const float*)d_in[13];
  const float* W2    = (const float*)d_in[14]; const float* b2 = (const float*)d_in[15];
  float* out = (float*)d_out;

  // workspace layout (floats); peak ~272 MB with reuse
  const size_t NBF = (size_t)NB_NODES * OUTd;     // 12.8M floats
  float* xf    = (float*)d_ws;
  float* q     = xf   + NBF;
  float* k     = q    + NBF;
  float* v     = k    + NBF;
  float* outb  = v    + NBF;
  float* alpha = outb + NBF;                      // T*H
  unsigned* keys = (unsigned*)(alpha + (size_t)T_EDGES * Hh);   // NB*H
  float* denom = (float*)keys + (size_t)NB_NODES * Hh;
  float* h1 = q;   // FFN hidden (NB x 256) overlays dead q+k
  float* h2 = v;   // FFN output (NB x 128) overlays dead v

  const dim3 tb(32, 8, 1);
  const dim3 tg((Nn + 31) / 32, Cdim / 32, Bg);
  transpose_x_kernel<<<tg, tb, 0, stream>>>(x, xf);

  auto gemm = [&](const float* X, const float* W, const float* bias, float* Y,
                  int M, int K, int Nc, int relu) {
    dim3 grid((M + 63) / 64, Nc / 128);
    wmma_gemm_kernel<<<grid, 256, 0, stream>>>(X, W, bias, Y, M, K, Nc, relu);
  };

  gemm(xf, Wq,    bq,    q,    NB_NODES, Cdim, OUTd, 0);
  gemm(xf, Wk,    bk,    k,    NB_NODES, Cdim, OUTd, 0);
  gemm(xf, Wv,    bv,    v,    NB_NODES, Cdim, OUTd, 0);
  gemm(xf, Wskip, bskip, outb, NB_NODES, Cdim, OUTd, 0);   // skip path seeds out accumulator

  hipMemsetAsync(keys,  0, (size_t)NB_NODES * Hh * sizeof(unsigned), stream);
  hipMemsetAsync(denom, 0, (size_t)NB_NODES * Hh * sizeof(float),    stream);

  edge_alpha_kernel<<<(T_EDGES + 7) / 8, 256, 0, stream>>>(eidx, eattr, We, q, k, alpha, keys);
  finalize_amax_kernel<<<(NB_NODES * Hh + 255) / 256, 256, 0, stream>>>(keys, NB_NODES * Hh);
  edge_softmax_kernel<<<(T_EDGES * Hh + 255) / 256, 256, 0, stream>>>(eidx, alpha, (const float*)keys, denom);
  edge_aggregate_kernel<<<(T_EDGES + 7) / 8, 256, 0, stream>>>(eidx, eattr, We, v, alpha, denom, outb);

  gemm(outb, W1, b1, h1, NB_NODES, OUTd, HIDd, 1);   // relu(out @ W1 + b1)
  gemm(h1,   W2, b2, h2, NB_NODES, HIDd, OUTd, 1);   // relu(.. @ W2 + b2)

  transpose_out_kernel<<<tg, tb, 0, stream>>>(h2, out);
}
